// LocalSelfAttention_59923383714548
// MI455X (gfx1250) — compile-verified, matrix operand order checked by reading, not run on hardware
//
#include <hip/hip_runtime.h>
#include <hip/hip_bf16.h>
#include <stdint.h>

// ---------------- problem constants ----------------
#define D_MODEL   1024
#define NHEAD     16
#define HEAD_DIM  64
#define SEQ_T     2048
#define BATCH     2
#define WIN       64
#define MTOT      (BATCH * SEQ_T)                                   // 4096
#define QKV_SEC   ((size_t)BATCH * NHEAD * SEQ_T * HEAD_DIM)        // 4,194,304 elems

typedef __attribute__((ext_vector_type(16))) __bf16 v16bf;
typedef __attribute__((ext_vector_type(8)))  float  v8f;

// fp32 -> bf16 bits, round-to-nearest-even
__device__ __forceinline__ unsigned short f2bf(float x) {
  union { float f; unsigned u; } c; c.f = x;
  unsigned r = (c.u + 0x7FFFu + ((c.u >> 16) & 1u)) >> 16;
  return (unsigned short)r;
}
__device__ __forceinline__ unsigned short to_bf_bits(float x)          { return f2bf(x); }
__device__ __forceinline__ unsigned short to_bf_bits(unsigned short x) { return x; }

__device__ __forceinline__ v8f wmma_bf16(v16bf a, v16bf b, v8f c) {
  // D(f32 16x16) = A(bf16 16x32) * B(bf16 32x16) + C
  return __builtin_amdgcn_wmma_f32_16x16x32_bf16(false, a, false, b, (short)0, c, false, false);
}

// async copy 16B global -> LDS (GLOBAL_LOAD_ASYNC_TO_LDS_B128, GV mode, ASYNCcnt)
__device__ __forceinline__ void async_cp16(const void* gsrc, void* lds_dst) {
  const unsigned lds = (unsigned)(uintptr_t)lds_dst;   // low 32 bits = LDS offset
  asm volatile("global_load_async_to_lds_b128 %0, %1, off"
               :: "v"(lds), "v"((unsigned long long)(uintptr_t)gsrc)
               : "memory");
}
__device__ __forceinline__ void async_wait0() {
  asm volatile("s_wait_asynccnt 0x0" ::: "memory");
}

// Fragment loader. Source memory holds bf16 bits as [row][k], row stride ldk (even).
// Per CDNA5 16-bit A layout (mirrored for B): lane half hl covers
// k = hl*8 + {0..7} and 16 + hl*8 + {0..7}, packed in bf16 pairs -> dword reads.
__device__ __forceinline__ v16bf load_frag(const unsigned short* base, int row, int ldk, int hl) {
  union { v16bf v; unsigned u[8]; } f;
  const unsigned* p = (const unsigned*)(base + row * ldk + hl * 8);
  #pragma unroll
  for (int g = 0; g < 2; ++g)
    #pragma unroll
    for (int jj = 0; jj < 4; ++jj)
      f.u[g * 4 + jj] = p[(g * 16 + jj * 2) >> 1];
  return f.v;
}

// ---------------- GEMM: out = A @ W + bias -----------------------------------
// 256 threads = 8 waves; block tile 128x128; wave tile 32x64 (2x4 WMMA tiles).
// MODE 0: A fp32 (x), write bf16 scattered into qkv ws laid out [3][B,H,T,Dh]
// MODE 1: A bf16 (attn ws, async-staged), write fp32 linear [M,N] to d_out
template <int MODE, typename AT>
__global__ __launch_bounds__(256)
void gemm_bf16_kernel(const AT* __restrict__ A, const float* __restrict__ W,
                      const float* __restrict__ bias, float* __restrict__ outF,
                      unsigned short* __restrict__ outQKV, int M, int N, int K) {
  __shared__ unsigned short Alds[128 * 40];   // [m][k], 80B rows (16B aligned)
  __shared__ unsigned short Blds[128 * 40];   // [n][k] transposed

  const int m0 = blockIdx.x * 128, n0 = blockIdx.y * 128;
  const int tid = threadIdx.x;
  const int w = tid >> 5, lane = tid & 31, hl = lane >> 4, nl = lane & 15;
  const int mSub = (w & 3) * 32, nSub = (w >> 2) * 64;

  v8f c[2][4];
  #pragma unroll
  for (int mi = 0; mi < 2; ++mi)
    #pragma unroll
    for (int ni = 0; ni < 4; ++ni)
      #pragma unroll
      for (int r = 0; r < 8; ++r) c[mi][ni][r] = 0.f;

  for (int k0 = 0; k0 < K; k0 += 32) {
    // ---- stage A panel 128x32 -> LDS ----
    if (sizeof(AT) == 2) {
      // bf16 source: direct async copy, 2 x B128 per thread (512 xfers total)
      #pragma unroll
      for (int j = 0; j < 2; ++j) {
        const int idx = tid + j * 256;
        const int r = idx >> 2, kq = (idx & 3) * 8;
        async_cp16(A + (size_t)(m0 + r) * K + k0 + kq, &Alds[r * 40 + kq]);
      }
    } else {
      const int r = tid >> 1, koff = (tid & 1) * 16;
      const AT* src = A + (size_t)(m0 + r) * K + k0 + koff;
      #pragma unroll
      for (int j = 0; j < 16; ++j) Alds[r * 40 + koff + j] = to_bf_bits(src[j]);
    }
    // ---- stage B panel 32x128 -> LDS transposed [n][k] (fp32 -> bf16) ----
    #pragma unroll
    for (int jj = 0; jj < 16; ++jj) {
      int linear = tid + jj * 256;           // coalesced over n
      int kk = linear >> 7, nn = linear & 127;
      Blds[nn * 40 + kk] = f2bf(W[(size_t)(k0 + kk) * N + n0 + nn]);
    }
    // prefetch next W panel into L2 (global_prefetch_b8)
    if (k0 + 32 < K)
      __builtin_prefetch(&W[(size_t)(k0 + 32 + (tid >> 3)) * N + n0 + (tid & 7) * 16], 0, 1);
    if (sizeof(AT) == 2) async_wait0();
    __syncthreads();

    v16bf a[2], b[4];
    #pragma unroll
    for (int mi = 0; mi < 2; ++mi) a[mi] = load_frag(Alds, mSub + mi * 16 + nl, 40, hl);
    #pragma unroll
    for (int ni = 0; ni < 4; ++ni) b[ni] = load_frag(Blds, nSub + ni * 16 + nl, 40, hl);
    #pragma unroll
    for (int mi = 0; mi < 2; ++mi)
      #pragma unroll
      for (int ni = 0; ni < 4; ++ni)
        c[mi][ni] = wmma_bf16(a[mi], b[ni], c[mi][ni]);
    __syncthreads();
  }

  // epilogue: C layout -> lane holds column n = nl, rows m = r + 8*hl per VGPR r
  #pragma unroll
  for (int mi = 0; mi < 2; ++mi)
    #pragma unroll
    for (int ni = 0; ni < 4; ++ni) {
      const int n = n0 + nSub + ni * 16 + nl;
      const float bv = bias[n];
      #pragma unroll
      for (int r = 0; r < 8; ++r) {
        const int m = m0 + mSub + mi * 16 + r + 8 * hl;
        const float val = c[mi][ni][r] + bv;
        if (MODE == 0) {
          const int s = n >> 10, cc = n & 1023, head = cc >> 6, d = cc & 63;
          const int b2 = m >> 11, t = m & 2047;
          outQKV[(size_t)s * QKV_SEC +
                 (((size_t)b2 * NHEAD + head) * SEQ_T + t) * HEAD_DIM + d] = f2bf(val);
        } else {
          outF[(size_t)m * N + n] = val;
        }
      }
    }
}

// ---------------- local attention (flash-style, window +/-64) ----------------
// Block = 4 waves = 64 queries for one (batch, head). Keys [q0-64, q0+127], 6 chunks of 32.
__global__ __launch_bounds__(128)
void local_attn_kernel(const unsigned short* __restrict__ qkv,
                       unsigned short* __restrict__ attn_out) {
  __shared__ unsigned short Klds[192 * 72];   // [key][d], 144B rows (16B aligned)
  __shared__ unsigned short Vlds[64 * 194];   // [d][key] (transposed)

  const int qblock = blockIdx.x, head = blockIdx.y, batch = blockIdx.z;
  const int bh = batch * NHEAD + head;
  const unsigned short* Q  = qkv + 0 * QKV_SEC + (size_t)bh * SEQ_T * HEAD_DIM;
  const unsigned short* Kp = qkv + 1 * QKV_SEC + (size_t)bh * SEQ_T * HEAD_DIM;
  const unsigned short* Vp = qkv + 2 * QKV_SEC + (size_t)bh * SEQ_T * HEAD_DIM;

  const int tid = threadIdx.x;
  const int kbase = qblock * 64 - WIN;

  // K panel: direct bf16 rows -> async copy to LDS (12 x B128 per thread)
  #pragma unroll
  for (int j = 0; j < 12; ++j) {
    const int idx = tid + j * 128;            // 0..1535
    const int row = idx >> 3, dq = (idx & 7) * 8;
    int g = kbase + row;
    g = g < 0 ? 0 : (g >= SEQ_T ? SEQ_T - 1 : g);   // clamp; masked later
    async_cp16(Kp + (size_t)g * HEAD_DIM + dq, &Klds[row * 72 + dq]);
  }
  // V panel: transposed store (no async transpose path)
  for (int j = tid; j < 192 * 64; j += 128) {
    const int row = j >> 6, d = j & 63;
    int g = kbase + row;
    g = g < 0 ? 0 : (g >= SEQ_T ? SEQ_T - 1 : g);
    Vlds[d * 194 + row] = Vp[(size_t)g * HEAD_DIM + d];
  }
  async_wait0();
  __syncthreads();

  const int w = tid >> 5, lane = tid & 31, hl = lane >> 4, nl = lane & 15;
  const int qrow = qblock * 64 + w * 16 + nl;         // this lane's query column

  // Q fragments (B operand of S^T = K * Q^T): rows = queries, k = d
  const unsigned short* qr = Q + (size_t)qrow * HEAD_DIM;
  const v16bf qb0 = load_frag(qr, 0, HEAD_DIM, hl);          // d 0..31
  const v16bf qb1 = load_frag(qr + 32, 0, HEAD_DIM, hl);     // d 32..63

  v8f o[4];
  #pragma unroll
  for (int dt = 0; dt < 4; ++dt)
    #pragma unroll
    for (int r = 0; r < 8; ++r) o[dt][r] = 0.f;
  float mrow = -1e30f, lrow = 0.f;

  for (int cc = 0; cc < 6; ++cc) {
    // S^T (keys x queries): two 16x16 C tiles, K-dim = 64 split in two WMMAs
    v8f s[2];
    #pragma unroll
    for (int kt = 0; kt < 2; ++kt) {
      const int keyrow = cc * 32 + kt * 16 + nl;      // A row = key index
      v16bf a0 = load_frag(Klds,      keyrow, 72, hl);
      v16bf a1 = load_frag(Klds + 32, keyrow, 72, hl);
      v8f z;
      #pragma unroll
      for (int r = 0; r < 8; ++r) z[r] = 0.f;
      v8f t0 = wmma_bf16(a0, qb0, z);
      s[kt]  = wmma_bf16(a1, qb1, t0);
    }

    // scale + window mask; per-query (lane) stats with one half-swap shuffle
    float sv[2][8], mx = -1e30f;
    #pragma unroll
    for (int kt = 0; kt < 2; ++kt)
      #pragma unroll
      for (int r = 0; r < 8; ++r) {
        const int gkey = kbase + cc * 32 + kt * 16 + r + 8 * hl;
        float x = s[kt][r] * 0.125f;                  // 1/sqrt(64)
        int dist = qrow - gkey; dist = dist < 0 ? -dist : dist;
        const bool ok = (gkey >= 0) && (gkey < SEQ_T) && (dist <= WIN);
        x = ok ? x : -1e30f;
        sv[kt][r] = x;
        mx = fmaxf(mx, x);
      }
    mx = fmaxf(mx, __shfl_xor(mx, 16, 32));
    const float mnew  = fmaxf(mrow, mx);
    const float alpha = __expf(mrow - mnew);

    float rs = 0.f;
    union { v16bf v; unsigned u[8]; } pf;             // P fragment: lane-local repack
    #pragma unroll
    for (int kt = 0; kt < 2; ++kt)
      #pragma unroll
      for (int r = 0; r < 8; r += 2) {
        const float p0 = __expf(sv[kt][r]     - mnew);
        const float p1 = __expf(sv[kt][r + 1] - mnew);
        rs += p0 + p1;
        pf.u[kt * 4 + (r >> 1)] = (unsigned)f2bf(p0) | ((unsigned)f2bf(p1) << 16);
      }
    rs += __shfl_xor(rs, 16, 32);
    lrow = lrow * alpha + rs;
    mrow = mnew;

    // broadcast alpha[query-row] to the O (d-column) layout, rescale, accumulate P*V
    float a8[8];
    #pragma unroll
    for (int r = 0; r < 8; ++r) a8[r] = __shfl(alpha, r + 8 * hl, 32);
    #pragma unroll
    for (int dt = 0; dt < 4; ++dt) {
      #pragma unroll
      for (int r = 0; r < 8; ++r) o[dt][r] *= a8[r];
      v16bf vb = load_frag(Vlds + cc * 32, dt * 16 + nl, 194, hl);
      o[dt] = wmma_bf16(pf.v, vb, o[dt]);
    }
  }

  // normalize and store bf16 attn-out as [B,T,C] with C = head*64 + d
  const float li = 1.f / lrow;
  float l8[8];
  #pragma unroll
  for (int r = 0; r < 8; ++r) l8[r] = __shfl(li, r + 8 * hl, 32);
  #pragma unroll
  for (int dt = 0; dt < 4; ++dt)
    #pragma unroll
    for (int r = 0; r < 8; ++r) {
      const int t = qblock * 64 + w * 16 + r + 8 * hl;
      attn_out[((size_t)batch * SEQ_T + t) * D_MODEL + head * HEAD_DIM + dt * 16 + nl] =
          f2bf(o[dt][r] * l8[r]);
    }
}

// ---------------- launch -----------------------------------------------------
extern "C" void kernel_launch(void* const* d_in, const int* in_sizes, int n_in,
                              void* d_out, int out_size, void* d_ws, size_t ws_size,
                              hipStream_t stream) {
  const float* x     = (const float*)d_in[0];
  const float* Wqkv  = (const float*)d_in[1];
  const float* bqkv  = (const float*)d_in[2];
  const float* Wproj = (const float*)d_in[3];
  const float* bproj = (const float*)d_in[4];
  float* out = (float*)d_out;

  // bf16 workspace: [3][B,H,T,Dh] qkv + [B,T,C] attn-out = 16M bf16 = 33.5 MB
  unsigned short* qkv_ws  = (unsigned short*)d_ws;
  unsigned short* attn_ws = qkv_ws + 3 * QKV_SEC;

  // 1) qkv = x @ W_qkv + b_qkv  (scatter to [3][B,H,T,Dh] bf16)
  gemm_bf16_kernel<0, float><<<dim3(MTOT / 128, (3 * D_MODEL) / 128), 256, 0, stream>>>(
      x, Wqkv, bqkv, nullptr, qkv_ws, MTOT, 3 * D_MODEL, D_MODEL);

  // 2) windowed attention per (q-block of 64, head, batch)
  local_attn_kernel<<<dim3(SEQ_T / 64, NHEAD, BATCH), 128, 0, stream>>>(qkv_ws, attn_ws);

  // 3) out = attn @ W_proj + b_proj  (fp32 output, A panel async-staged)
  gemm_bf16_kernel<1, unsigned short><<<dim3(MTOT / 128, D_MODEL / 128), 256, 0, stream>>>(
      attn_ws, Wproj, bproj, out, nullptr, MTOT, D_MODEL, D_MODEL);
}
